// GCNLayer_11879879544543
// MI455X (gfx1250) — compile-verified, matrix operand order checked by reading, not run on hardware
//
#include <hip/hip_runtime.h>

// ---------- types ----------
typedef __attribute__((ext_vector_type(16))) __bf16 v16bf;
typedef __attribute__((ext_vector_type(8)))  __bf16 v8bf;
typedef __attribute__((ext_vector_type(8)))  float  v8f;

// ---------- problem constants ----------
#define BDIM   16
#define CIN    32
#define NDIM   2048
#define TDIM   12
#define KCH    3
#define COUT   32
#define GJ     6144          // j = (b*32+o)*12 + t
#define GK     6144          // q = k*2048 + m
#define GN     2048          // n
#define OUT_TOTAL (BDIM*COUT*NDIM*TDIM)   // 12,582,912
#define PLANE  (NDIM*TDIM)   // 24576 elements per (b,o) plane
#define BN_EPS 1e-5f

// ---------- GEMM tiling ----------
#define BM  128
#define BNT 128
#define BK  64               // deep k-tile: 48 WMMAs per sync interval
#define LDT 72               // padded LDS row stride (elems): 144B rows -> 16B-aligned, conflict-free
#define NKT (GK / BK)        // 96 k-tiles

// =====================================================================
// Kernel 1: tiled transpose+split of cheb [k][m][n] -> chebT hi/lo [n][q]
// grid (64, 64, 3), 256 threads, 32x32 tiles
// =====================================================================
__global__ void __launch_bounds__(256)
transpose_cheb_kernel(const float* __restrict__ cheb,
                      __bf16* __restrict__ th, __bf16* __restrict__ tl) {
  __shared__ float tile[32][33];
  const int tid = threadIdx.x;
  const int tx = tid & 31, ty = tid >> 5;          // ty 0..7
  const int n0 = blockIdx.x * 32;
  const int m0 = blockIdx.y * 32;
  const int k  = blockIdx.z;
#pragma unroll
  for (int i = 0; i < 4; ++i) {
    const int m = ty + i * 8;
    tile[m][tx] = cheb[((size_t)(k * NDIM + m0 + m)) * NDIM + n0 + tx];  // coalesced in n
  }
  __syncthreads();
#pragma unroll
  for (int i = 0; i < 4; ++i) {
    const int n = ty + i * 8;
    const float v = tile[tx][n];
    const __bf16 hv = (__bf16)v;
    const size_t o = (size_t)(n0 + n) * GK + (size_t)k * NDIM + m0 + tx;  // coalesced in m
    th[o] = hv;
    tl[o] = (__bf16)(v - (float)hv);
  }
}

// =====================================================================
// Kernel 2: s_hi/lo[j][q] = split( sum_c relu(x[b,c,m,t]) * theta[k,c,o] )
// grid: 16*12*8 = 1536 blocks of 256 threads (one m per thread)
// =====================================================================
__global__ void __launch_bounds__(256)
prep_s_kernel(const float* __restrict__ x, const float* __restrict__ theta,
              __bf16* __restrict__ sh, __bf16* __restrict__ sl) {
  __shared__ float thetaT[KCH * COUT * CIN];   // [k][o][c], c contiguous
  const int tid = threadIdx.x;
  const int blk = blockIdx.x;
  const int mc = blk & 7;
  const int t  = (blk >> 3) % TDIM;
  const int b  = blk / (8 * TDIM);

  for (int i = tid; i < KCH * COUT * CIN; i += 256) {
    const int c = i & 31, o = (i >> 5) & 31, k = i >> 10;
    thetaT[(k * 32 + o) * 32 + c] = theta[(k * 32 + c) * 32 + o];
  }

  const int m = mc * 256 + tid;
  float xv[CIN];
#pragma unroll
  for (int c = 0; c < CIN; ++c) {
    const float v = x[(((size_t)(b * 32 + c)) * NDIM + m) * TDIM + t];
    xv[c] = v > 0.f ? v : 0.f;
  }
  __syncthreads();

  const int jbase = (b * 32) * TDIM + t;
  for (int k = 0; k < KCH; ++k) {
    for (int o = 0; o < COUT; ++o) {
      const float* tc = &thetaT[(k * 32 + o) * 32];
      float acc = 0.f;
#pragma unroll
      for (int c = 0; c < CIN; ++c) acc += xv[c] * tc[c];
      const int j = jbase + o * TDIM;
      const size_t idx = (size_t)j * GK + k * NDIM + m;   // coalesced over m
      const __bf16 hv = (__bf16)acc;
      sh[idx] = hv;
      sl[idx] = (__bf16)(acc - (float)hv);
    }
  }
}

// =====================================================================
// Fragment load: 16x32 bf16 A-layout (B symmetric) from LDS row-major
// lane l: row = l&15, half h = l>>4, K = {8h..8h+7} U {16+8h..16+8h+7}
// =====================================================================
__device__ __forceinline__ v16bf ld_frag(const __bf16* base, int row0, int ksub, int lane) {
  const int r = lane & 15, h = lane >> 4;
  const __bf16* p = base + (row0 + r) * LDT + ksub + 8 * h;   // 16B aligned
  const v8bf a = *(const v8bf*)p;
  const v8bf b = *(const v8bf*)(p + 16);
  return __builtin_shufflevector(a, b, 0, 1, 2, 3, 4, 5, 6, 7,
                                 8, 9, 10, 11, 12, 13, 14, 15);
}

// =====================================================================
// Kernel 3: out[n][j] = sum_q chebT[n][q] * s[q][j], split-bf16 (3 WMMA/tile)
// grid (16, 48), 256 threads = 8 waves (2x4), wave tile 64n x 32j.
// Double-buffered staging via global_load_async_to_lds_b128 (ASYNCcnt).
// =====================================================================
__global__ void __launch_bounds__(256)
gemm_kernel(const __bf16* __restrict__ ah_g, const __bf16* __restrict__ al_g,
            const __bf16* __restrict__ bh_g, const __bf16* __restrict__ bl_g,
            float* __restrict__ out) {
  __shared__ __align__(16) __bf16 Ah[2][BM * LDT], Al[2][BM * LDT];
  __shared__ __align__(16) __bf16 Bh[2][BNT * LDT], Bl[2][BNT * LDT];

  const int tid  = threadIdx.x;
  const int lane = tid & 31;
  const int wid  = tid >> 5;
  const int wm   = wid >> 2;      // 0..1 -> 64 n-rows
  const int wn   = wid & 3;       // 0..3 -> 32 j-cols
  const int n0   = blockIdx.x * BM;
  const int j0   = blockIdx.y * BNT;

  // ---- async staging ownership: wave -> one of 4 arrays, half of 128 rows ----
  // Per stage: 4 arrays x 128 rows x 128B = 64KB = 128 b128 ops = 16 per wave.
  const int arr   = wid >> 1;            // 0:Ah 1:Al 2:Bh 3:Bl
  const int rhalf = (wid & 1) * 64;      // row half
  const int rsub  = lane >> 3;           // 0..3 row within 4-row group
  const int csub  = (lane & 7) * 8;      // 8-elem (16B) chunk within 128B row
  const __bf16* gsrc;
  __bf16* lb[2];
  int rbase;
  if (arr == 0)      { gsrc = ah_g; lb[0] = Ah[0]; lb[1] = Ah[1]; rbase = n0; }
  else if (arr == 1) { gsrc = al_g; lb[0] = Al[0]; lb[1] = Al[1]; rbase = n0; }
  else if (arr == 2) { gsrc = bh_g; lb[0] = Bh[0]; lb[1] = Bh[1]; rbase = j0; }
  else               { gsrc = bl_g; lb[0] = Bl[0]; lb[1] = Bl[1]; rbase = j0; }

  auto issue_stage = [&](int kt, int buf) {
#pragma unroll
    for (int i = 0; i < 16; ++i) {
      const int row = rhalf + i * 4 + rsub;
      const __bf16* g = gsrc + (size_t)(rbase + row) * GK + (size_t)kt * BK + csub;
      const unsigned lds = (unsigned)(unsigned long)(lb[buf] + row * LDT + csub);
      asm volatile("global_load_async_to_lds_b128 %0, %1, off"
                   :: "v"(lds), "v"(g) : "memory");
    }
  };

  v8f acc[4][2] = {};

  issue_stage(0, 0);
  for (int kt = 0; kt < NKT; ++kt) {
    const int cb = kt & 1;
    if (kt + 1 < NKT) {
      issue_stage(kt + 1, cb ^ 1);
      asm volatile("s_wait_asynccnt 0x10" ::: "memory");  // tile kt complete (in-order)
    } else {
      asm volatile("s_wait_asynccnt 0x0" ::: "memory");
    }
    __syncthreads();   // all waves' async writes for tile kt visible

    // ---- compute: 2 k-substeps x 8 m/n tiles x 3 split-bf16 WMMAs ----
#pragma unroll
    for (int ks = 0; ks < BK; ks += 32) {
      v16bf bh[2], bl[2];
#pragma unroll
      for (int jn = 0; jn < 2; ++jn) {
        bh[jn] = ld_frag(Bh[cb], wn * 32 + jn * 16, ks, lane);
        bl[jn] = ld_frag(Bl[cb], wn * 32 + jn * 16, ks, lane);
      }
#pragma unroll
      for (int im = 0; im < 4; ++im) {
        const v16bf ah = ld_frag(Ah[cb], wm * 64 + im * 16, ks, lane);
        const v16bf al = ld_frag(Al[cb], wm * 64 + im * 16, ks, lane);
#pragma unroll
        for (int jn = 0; jn < 2; ++jn) {
          acc[im][jn] = __builtin_amdgcn_wmma_f32_16x16x32_bf16(
              false, ah, false, bh[jn], (short)0, acc[im][jn], false, false);
          acc[im][jn] = __builtin_amdgcn_wmma_f32_16x16x32_bf16(
              false, ah, false, bl[jn], (short)0, acc[im][jn], false, false);
          acc[im][jn] = __builtin_amdgcn_wmma_f32_16x16x32_bf16(
              false, al, false, bh[jn], (short)0, acc[im][jn], false, false);
        }
      }
    }
    __syncthreads();   // all waves done reading buffer cb before it is re-staged
  }

  // ---- epilogue: D frag (lane: col=l&15, row=v+8*(l>>4)) -> out[b,o,n,t] ----
  const int hh  = lane >> 4;
  const int col = lane & 15;
#pragma unroll
  for (int im = 0; im < 4; ++im) {
    const int nb = n0 + wm * 64 + im * 16;
#pragma unroll
    for (int jn = 0; jn < 2; ++jn) {
      const int j = j0 + wn * 32 + jn * 16 + col;        // (b*32+o)*12 + t
      const int head = (j / TDIM) * PLANE + (j % TDIM);
#pragma unroll
      for (int v = 0; v < 8; ++v) {
        const int n = nb + v + 8 * hh;
        out[(size_t)head + (size_t)n * TDIM] = acc[im][jn][v];
      }
    }
  }
}

// =====================================================================
// Kernel 4: per-(b,o)-plane partial sums (deterministic, no float atomics)
// =====================================================================
__global__ void __launch_bounds__(256)
stats_kernel(const float* __restrict__ out, float* __restrict__ partial) {
  __shared__ float rs[256], rq[256];
  const int tid = threadIdx.x;
  const int plane = blockIdx.x;           // 0..511 = b*32+o
  const float* p = out + (size_t)plane * PLANE;
  float sm = 0.f, sq = 0.f;
  for (int i = tid; i < PLANE; i += 256) {
    const float v = p[i];
    sm += v; sq += v * v;
  }
  rs[tid] = sm; rq[tid] = sq;
  __syncthreads();
  for (int off = 128; off > 0; off >>= 1) {
    if (tid < off) { rs[tid] += rs[tid + off]; rq[tid] += rq[tid + off]; }
    __syncthreads();
  }
  if (tid == 0) { partial[plane * 2] = rs[0]; partial[plane * 2 + 1] = rq[0]; }
}

__global__ void reduce_stats_kernel(const float* __restrict__ partial,
                                    float* __restrict__ stats) {
  const int o = threadIdx.x;
  if (o < COUT) {
    float sm = 0.f, sq = 0.f;
    for (int b = 0; b < BDIM; ++b) {
      sm += partial[(b * 32 + o) * 2];
      sq += partial[(b * 32 + o) * 2 + 1];
    }
    stats[o] = sm;
    stats[32 + o] = sq;
  }
}

// =====================================================================
// Kernel 5: in-place normalize (biased var, matches jnp.var)
// =====================================================================
__global__ void __launch_bounds__(256)
bn_kernel(float* __restrict__ out, const float* __restrict__ stats) {
  __shared__ float smean[COUT], sscale[COUT];
  const int tid = threadIdx.x;
  if (tid < COUT) {
    const float inv = 1.0f / (float)(BDIM * NDIM * TDIM);   // 1/393216
    const float m = stats[tid] * inv;
    const float var = stats[32 + tid] * inv - m * m;
    smean[tid] = m;
    sscale[tid] = rsqrtf(var + BN_EPS);
  }
  __syncthreads();
  for (int i = blockIdx.x * 256 + tid; i < OUT_TOTAL; i += gridDim.x * 256) {
    const int o = (i / PLANE) & 31;
    out[i] = (out[i] - smean[o]) * sscale[o];
  }
}

// =====================================================================
extern "C" void kernel_launch(void* const* d_in, const int* in_sizes, int n_in,
                              void* d_out, int out_size, void* d_ws, size_t ws_size,
                              hipStream_t stream) {
  const float* x     = (const float*)d_in[0];   // [16,32,2048,12]
  const float* cheb  = (const float*)d_in[1];   // [3,2048,2048]
  const float* theta = (const float*)d_in[2];   // [3,32,32]
  float* out = (float*)d_out;                   // [16,32,2048,12]

  // workspace layout (bf16 operand arrays + BN scratch)
  const size_t chebT_elems = (size_t)GN * GK;   // 12,582,912
  const size_t s_elems     = (size_t)GJ * GK;   // 37,748,736
  __bf16* chebT_hi = (__bf16*)d_ws;
  __bf16* chebT_lo = chebT_hi + chebT_elems;
  __bf16* s_hi     = chebT_lo + chebT_elems;
  __bf16* s_lo     = s_hi + s_elems;
  float*  partial  = (float*)(s_lo + s_elems);  // 1024 floats
  float*  stats    = partial + 1024;            // 64 floats

  dim3 tg(GN / 32, GN / 32, KCH);  // (64,64,3)
  transpose_cheb_kernel<<<tg, 256, 0, stream>>>(cheb, chebT_hi, chebT_lo);

  prep_s_kernel<<<BDIM * TDIM * (NDIM / 256), 256, 0, stream>>>(x, theta, s_hi, s_lo);

  dim3 g(GN / BM, GJ / BNT);       // (16, 48)
  gemm_kernel<<<g, 256, 0, stream>>>(chebT_hi, chebT_lo, s_hi, s_lo, out);

  stats_kernel<<<BDIM * COUT, 256, 0, stream>>>(out, partial);
  reduce_stats_kernel<<<1, 64, 0, stream>>>(partial, stats);
  bn_kernel<<<1024, 256, 0, stream>>>(out, stats);
}
